// HYPERGC_73461120631196
// MI455X (gfx1250) — compile-verified, matrix operand order checked by reading, not run on hardware
//
#include <hip/hip_runtime.h>
#include <hip/hip_bf16.h>

typedef _Float16 v16h __attribute__((ext_vector_type(16)));
typedef _Float16 v8h  __attribute__((ext_vector_type(8)));
typedef float    v8f  __attribute__((ext_vector_type(8)));

#define NN 64
#define CC 64
#define TT 512
#define VV 25
#define SS 8
#define HD 16
#define CING 8

// ---------------------------------------------------------------------------
// K1: t_x[n,c,v] = mean over T of x[n,c,t,v].  Block per (n,c).
// ---------------------------------------------------------------------------
__global__ void __launch_bounds__(256) k1_mean(const float* __restrict__ x,
                                               float* __restrict__ tx) {
  const int nc  = blockIdx.x;          // n*CC + c
  const int tid = threadIdx.x;
  const int g   = tid >> 5;            // 0..7 t-groups
  const int v   = tid & 31;
  __shared__ float part[8][32];
  float acc = 0.f;
  if (v < VV) {
    const float* base = x + (size_t)nc * TT * VV;
    for (int t = g; t < TT; t += 8) acc += base[t * VV + v];
  }
  part[g][v] = acc;
  __syncthreads();
  if (g == 0 && v < VV) {
    float s = 0.f;
#pragma unroll
    for (int gg = 0; gg < 8; ++gg) s += part[gg][v];
    tx[(size_t)nc * VV + v] = s * (1.0f / (float)TT);
  }
}

// ---------------------------------------------------------------------------
// K2: per-n attention graph -> A_fused[n][v][u] (written into d_out tail).
// ---------------------------------------------------------------------------
#define OFF_TX 0        // 1600  t_x [64][25]
#define OFF_QS 1600     // 3200  Q   [8][25][16]
#define OFF_KS 4800     // 3200  K   [8][25][16]
#define OFF_AH 8000     // 5000  A_h [8][25][25]  (also reused as w1 [128][25])
#define OFF_W2 13000    // 200   w2  [8][25]
#define OFF_WR 13200    // 8
#define OFF_OM 13208    // 8
#define OFF_HS 13216    // 625   H_sem
#define OFF_RS 13841    // 25    row sums H_sem
#define OFF_RL 13866    // 25    row sums edge*PA
#define SM_FLOATS 13891

__global__ void __launch_bounds__(256) k2_attn(
    const float* __restrict__ tx,
    const float* __restrict__ Wq, const float* __restrict__ bq,
    const float* __restrict__ Wk, const float* __restrict__ bk,
    const float* __restrict__ Ww1, const float* __restrict__ bw1,
    const float* __restrict__ Ww2, const float* __restrict__ bw2,
    const float* __restrict__ conf_gate, const float* __restrict__ alpha,
    const float* __restrict__ PA, const float* __restrict__ edge,
    float* __restrict__ Af) {
  __shared__ float sm[SM_FLOATS];
  const int n = blockIdx.x;
  const int tid = threadIdx.x;

  for (int i = tid; i < CC * VV; i += 256) sm[OFF_TX + i] = tx[(size_t)n * CC * VV + i];
  __syncthreads();

  // w1 = leaky_relu(grouped conv Ww1), [128][25], stored in AH region
  for (int i = tid; i < SS * HD * VV; i += 256) {
    int o = i / VV, v = i % VV;
    int s = o >> 4;
    float acc = bw1[o];
#pragma unroll
    for (int k = 0; k < CING; ++k) acc += Ww1[o * CING + k] * sm[OFF_TX + (s * CING + k) * VV + v];
    sm[OFF_AH + o * VV + v] = acc > 0.f ? acc : 0.01f * acc;
  }
  __syncthreads();

  // w2 = tanh(Ww2 @ w1 + bw2), [8][25]
  if (tid < SS * VV) {
    int s = tid / VV, v = tid % VV;
    float acc = bw2[s];
    for (int c = 0; c < SS * HD; ++c) acc += sm[OFF_AH + c * VV + v] * Ww2[s * SS * HD + c];
    sm[OFF_W2 + tid] = tanhf(acc);
  }
  __syncthreads();
  if (tid < SS) {
    float m = 0.f;
#pragma unroll
    for (int v = 0; v < VV; ++v) m += sm[OFF_W2 + tid * VV + v];
    sm[OFF_WR + tid] = m * (1.0f / (float)VV);
  }
  __syncthreads();
  if (tid == 0) {  // omega = softmax_s(conf_gate + W_raw)
    float mx = -1e30f;
    for (int s = 0; s < SS; ++s) {
      float z = conf_gate[s] + sm[OFF_WR + s];
      mx = z > mx ? z : mx;
    }
    float sum = 0.f;
    for (int s = 0; s < SS; ++s) {
      float e = expf(conf_gate[s] + sm[OFF_WR + s] - mx);
      sm[OFF_OM + s] = e;
      sum += e;
    }
    float inv = 1.f / sum;
    for (int s = 0; s < SS; ++s) sm[OFF_OM + s] *= inv;
  }
  __syncthreads();

  // Q, K grouped convs: [8][25][16]
  for (int i = tid; i < SS * VV * HD; i += 256) {
    int s = i / (VV * HD), rem = i % (VV * HD);
    int v = rem / HD, d = rem % HD;
    int o = s * HD + d;
    float q = bq[o], k = bk[o];
#pragma unroll
    for (int c = 0; c < CING; ++c) {
      float tv = sm[OFF_TX + (s * CING + c) * VV + v];
      q += Wq[o * CING + c] * tv;
      k += Wk[o * CING + c] * tv;
    }
    sm[OFF_QS + (s * VV + v) * HD + d] = q;
    sm[OFF_KS + (s * VV + v) * HD + d] = k;
  }
  __syncthreads();

  // A_h[s][v][u] = scale * Q.K   (overwrites w1 region)
  for (int i = tid; i < SS * VV * VV; i += 256) {
    int s = i / (VV * VV), rem = i % (VV * VV);
    int v = rem / VV, u = rem % VV;
    float acc = 0.f;
#pragma unroll
    for (int d = 0; d < HD; ++d)
      acc += sm[OFF_QS + (s * VV + v) * HD + d] * sm[OFF_KS + (s * VV + u) * HD + d];
    sm[OFF_AH + i] = acc * 0.25f;  // HD^-0.5
  }
  __syncthreads();

  // top-9 mask + masked softmax + omega gating, per row (s,v)
  if (tid < SS * VV) {
    int s = tid / VV;
    float* row = &sm[OFF_AH + tid * VV];
    float buf[VV], ebuf[VV];
#pragma unroll
    for (int u = 0; u < VV; ++u) buf[u] = row[u];
    unsigned sel = 0;
    float m0 = -1e30f;
    for (int k = 0; k < 9; ++k) {
      float best = -1e30f;
      int bi = 0;
      for (int u = 0; u < VV; ++u)
        if (!((sel >> u) & 1u) && buf[u] > best) { best = buf[u]; bi = u; }
      sel |= (1u << bi);
      if (k == 0) m0 = best;
    }
    float sum = 0.f;
    for (int u = 0; u < VV; ++u) {
      float e = ((sel >> u) & 1u) ? expf(buf[u] - m0) : 0.f;
      ebuf[u] = e;
      sum += e;
    }
    float sc = sm[OFF_OM + s] / sum;
    for (int u = 0; u < VV; ++u) row[u] = ebuf[u] * sc;
  }
  __syncthreads();

  if (tid < VV * VV) {
    float h = 0.f;
#pragma unroll
    for (int s = 0; s < SS; ++s) h += sm[OFF_AH + s * VV * VV + tid];
    sm[OFF_HS + tid] = h;
  }
  __syncthreads();
  if (tid < VV) {
    float rs = 0.f, rl = 0.f;
    for (int u = 0; u < VV; ++u) {
      rs += fabsf(sm[OFF_HS + tid * VV + u]);
      rl += fabsf(edge[tid * VV + u] * PA[tid * VV + u]);
    }
    sm[OFF_RS + tid] = rs;
    sm[OFF_RL + tid] = rl;
  }
  __syncthreads();
  if (tid < VV * VV) {
    int v = tid / VV;
    float al = edge[tid] * PA[tid] / (sm[OFF_RL + v] + 1e-8f);
    float as = sm[OFF_HS + tid] / (sm[OFF_RS + v] + 1e-8f);
    float a = alpha[0] > 0.f ? alpha[0] : 0.f;
    Af[(size_t)n * VV * VV + tid] = al + a * as;
  }
}

// ---------------------------------------------------------------------------
// Fused GEMMs via WMMA f16 with fragment-order LDS layouts.
// ---------------------------------------------------------------------------
// Block involution on a 32-wide K block: [0..7,8..15,16..23,24..31] ->
// [0..7,16..23,8..15,24..31].  With this permutation, lane-group g's 16
// fragment elements are 16 contiguous halves starting at position g*16.
__device__ __forceinline__ int swap32(int q) {
  int blk = (q >> 3) & 3;
  blk = (blk == 1) ? 2 : (blk == 2) ? 1 : blk;
  return (blk << 3) | (q & 7);
}

__device__ __forceinline__ v16h ldfrag(const _Float16* p) {
  union { v16h f; v8h h[2]; } u;
  u.h[0] = *(const v8h*)(p);
  u.h[1] = *(const v8h*)(p + 8);
  return u.f;
}

template <bool WRITE>
__global__ void __launch_bounds__(256) k_gemm(
    const float* __restrict__ x, const float* __restrict__ Wd,
    const float* __restrict__ bd, const float* __restrict__ Af,
    const float* __restrict__ scale, const float* __restrict__ shift,
    float* __restrict__ gsum, float* __restrict__ gsq,
    float* __restrict__ out) {
  constexpr int TB = 4;                      // time steps per block
  constexpr int WH_S = 72;                   // padded strides (halves)
  constexpr int XP_S = 72;
  constexpr int DS_S = 136;
  constexpr int A2_S = 40;
  __shared__ _Float16 Wh[64 * WH_S];         // Wd, fragment order
  __shared__ _Float16 Xp[TB * 32 * XP_S];    // X, K-major per column
  __shared__ _Float16 Ds[64 * DS_S];         // GEMM1 result, fragment order
  __shared__ _Float16 A2[32 * A2_S];         // A_fused^T, K-major per column
  __shared__ float ssum[64], ssq[64];

  const int n  = blockIdx.x / (TT / TB);
  const int t0 = (blockIdx.x % (TT / TB)) * TB;
  const int tid = threadIdx.x;

  for (int i = tid; i < 64 * 64; i += 256) {
    int row = i >> 6, p = i & 63;
    int k = (p & 32) | swap32(p & 31);
    Wh[row * WH_S + p] = (_Float16)Wd[row * 64 + k];
  }
  for (int i = tid; i < TB * 64 * 32; i += 256) {
    int v = i & 31, c = (i >> 5) & 63, tt = i >> 11;
    float val = (v < VV) ? x[(((size_t)n * CC + c) * TT + (t0 + tt)) * VV + v] : 0.f;
    int p = (c & 32) | swap32(c & 31);
    Xp[(tt * 32 + v) * XP_S + p] = (_Float16)val;
  }
  for (int i = tid; i < 32 * 32; i += 256) {
    int u = i >> 5, q = i & 31;
    int v = swap32(q);
    float val = (u < VV && v < VV) ? Af[(size_t)n * VV * VV + u * VV + v] : 0.f;
    A2[u * A2_S + q] = (_Float16)val;
  }
  if (!WRITE && tid < 64) { ssum[tid] = 0.f; ssq[tid] = 0.f; }
  __syncthreads();

  const int wave = tid >> 5, lane = tid & 31;
  const int mi = wave >> 1, ni = wave & 1;   // 4x2 tiles of 16x16 per tt
  const int g = lane >> 4, r = lane & 15;
  const int col = ni * 16 + r;
  const int pcol = swap32(col);

  // GEMM1: D = Wd @ X + bd  (M=64, N=32 per tt, K=64 -> 2 WMMA per tile)
  for (int tt = 0; tt < TB; ++tt) {
    v8f acc;
#pragma unroll
    for (int i = 0; i < 8; ++i) acc[i] = bd[mi * 16 + 8 * g + i];
#pragma unroll
    for (int kk = 0; kk < 2; ++kk) {
      v16h a = ldfrag(&Wh[(mi * 16 + r) * WH_S + kk * 32 + g * 16]);
      v16h b = ldfrag(&Xp[(tt * 32 + col) * XP_S + kk * 32 + g * 16]);
      acc = __builtin_amdgcn_wmma_f32_16x16x32_f16(false, a, false, b, (short)0,
                                                   acc, false, false);
    }
#pragma unroll
    for (int i = 0; i < 8; ++i) {
      int row = mi * 16 + 8 * g + i;
      Ds[row * DS_S + tt * 32 + pcol] =
          (col < VV) ? (_Float16)acc[i] : (_Float16)0.f;  // kill bias in pad
    }
  }
  __syncthreads();

  // GEMM2: Y = D @ A^T  (K = v padded to 32, 1 WMMA per tile)
  v16h b2 = ldfrag(&A2[col * A2_S + g * 16]);
  float lsum[8], lsq[8];
#pragma unroll
  for (int i = 0; i < 8; ++i) { lsum[i] = 0.f; lsq[i] = 0.f; }

  for (int tt = 0; tt < TB; ++tt) {
    v16h a2 = ldfrag(&Ds[(mi * 16 + r) * DS_S + tt * 32 + g * 16]);
    v8f y = {};
    y = __builtin_amdgcn_wmma_f32_16x16x32_f16(false, a2, false, b2, (short)0,
                                               y, false, false);
    if (WRITE) {
      if (col < VV) {
#pragma unroll
        for (int i = 0; i < 8; ++i) {
          int row = mi * 16 + 8 * g + i;
          size_t idx = (((size_t)n * CC + row) * TT + (t0 + tt)) * VV + col;
          float val = y[i] * scale[row] + shift[row] + x[idx];
          out[idx] = val > 0.f ? val : 0.f;
        }
      }
    } else {
      if (col < VV) {
#pragma unroll
        for (int i = 0; i < 8; ++i) { lsum[i] += y[i]; lsq[i] += y[i] * y[i]; }
      }
    }
  }

  if (!WRITE) {
    if (col < VV) {
#pragma unroll
      for (int i = 0; i < 8; ++i) {
        int row = mi * 16 + 8 * g + i;
        atomicAdd(&ssum[row], lsum[i]);
        atomicAdd(&ssq[row], lsq[i]);
      }
    }
    __syncthreads();
    if (tid < 64) {
      atomicAdd(&gsum[tid], ssum[tid]);
      atomicAdd(&gsq[tid], ssq[tid]);
    }
  }
}

// ---------------------------------------------------------------------------
// K4: finalize BN batch stats into per-channel scale/shift.
// ---------------------------------------------------------------------------
__global__ void k_final(const float* __restrict__ gsum, const float* __restrict__ gsq,
                        const float* __restrict__ bn_w, const float* __restrict__ bn_b,
                        float* __restrict__ scale, float* __restrict__ shift) {
  int c = threadIdx.x;
  if (c < CC) {
    const float cnt = (float)NN * (float)TT * (float)VV;
    float mu = gsum[c] / cnt;
    float var = gsq[c] / cnt - mu * mu;
    float inv = rsqrtf(var + 1e-5f);
    float sc = bn_w[c] * inv;
    scale[c] = sc;
    shift[c] = bn_b[c] - mu * sc;
  }
}

// ---------------------------------------------------------------------------
extern "C" void kernel_launch(void* const* d_in, const int* in_sizes, int n_in,
                              void* d_out, int out_size, void* d_ws, size_t ws_size,
                              hipStream_t stream) {
  const float* x     = (const float*)d_in[0];
  const float* Wq    = (const float*)d_in[1];
  const float* bq    = (const float*)d_in[2];
  const float* Wk    = (const float*)d_in[3];
  const float* bk    = (const float*)d_in[4];
  const float* Ww1   = (const float*)d_in[5];
  const float* bw1   = (const float*)d_in[6];
  const float* Ww2   = (const float*)d_in[7];
  const float* bw2   = (const float*)d_in[8];
  const float* conf  = (const float*)d_in[9];
  const float* alpha = (const float*)d_in[10];
  const float* PA    = (const float*)d_in[11];
  const float* edge  = (const float*)d_in[12];
  const float* Wd    = (const float*)d_in[13];
  const float* bd    = (const float*)d_in[14];
  const float* bn_w  = (const float*)d_in[15];
  const float* bn_b  = (const float*)d_in[16];

  float* out = (float*)d_out;
  float* Af  = out + (size_t)NN * CC * TT * VV;  // second tuple output: A_fused

  float* ws    = (float*)d_ws;
  float* gsum  = ws;        // 64
  float* gsq   = ws + 64;   // 64
  float* scale = ws + 128;  // 64
  float* shift = ws + 192;  // 64
  float* tx    = ws + 256;  // N*C*V

  hipMemsetAsync(gsum, 0, 128 * sizeof(float), stream);

  k1_mean<<<NN * CC, 256, 0, stream>>>(x, tx);
  k2_attn<<<NN, 256, 0, stream>>>(tx, Wq, bq, Wk, bk, Ww1, bw1, Ww2, bw2,
                                  conf, alpha, PA, edge, Af);
  k_gemm<false><<<NN * (TT / 4), 256, 0, stream>>>(x, Wd, bd, Af, scale, shift,
                                                   gsum, gsq, out);
  k_final<<<1, 64, 0, stream>>>(gsum, gsq, bn_w, bn_b, scale, shift);
  k_gemm<true><<<NN * (TT / 4), 256, 0, stream>>>(x, Wd, bd, Af, scale, shift,
                                                  gsum, gsq, out);
}
